// TCN_GCN_unit_16922171146682
// MI455X (gfx1250) — matrix operand order repaired and checked by reading, not compiled, not run on hardware
//
#include <hip/hip_runtime.h>

// ---------------------------------------------------------------------------
// CTR-GCN block for MI455X (gfx1250, wave32).  bf16 WMMA (16x16x32) for all
// GEMM-shaped stages, VALU for the small attention math / temporal convs.
// B operands stored [n][k] so every WMMA operand load is 2x 16B vector load.
// Stage-2 accumulates across the 3 graph subsets in the WMMA C operand.
// ---------------------------------------------------------------------------

#define N_  128
#define C_  64
#define T_  128
#define V_  25
#define S_  3
#define R_  8
#define O_  64
#define K_  5
#define BC_ 16
#define AC_ 16

typedef __attribute__((ext_vector_type(16))) __bf16 v16bf;
typedef __attribute__((ext_vector_type(8)))  __bf16 v8bf;
typedef __attribute__((ext_vector_type(8)))  float  v8f;

__device__ __forceinline__ __bf16 f2b(float f) { return (__bf16)f; }
__device__ __forceinline__ float  b2f(__bf16 b) { return (float)b; }

// ---- WMMA per-lane operand loaders (CDNA5 ISA 7.12.2 layouts) --------
// A: 16x32 (MxK) 16-bit, stored row-major [m][k].  lane m = lane&15;
// lanes<16 hold K {0..7,16..23}, lanes>=16 hold K {8..15,24..31}.
__device__ __forceinline__ v16bf load_A(const __bf16* base, int stride, int lane) {
  int m  = lane & 15;
  int kb = (lane < 16) ? 0 : 8;
  const __bf16* p = base + m * stride + kb;
  v8bf lo = *(const v8bf*)(p);
  v8bf hi = *(const v8bf*)(p + 16);
  return __builtin_shufflevector(lo, hi, 0, 1, 2, 3, 4, 5, 6, 7,
                                 8, 9, 10, 11, 12, 13, 14, 15);
}
// B: 32x16 (KxN) 16-bit, stored TRANSPOSED [n][k].  lane n = lane&15;
// lanes<16 hold K {0..7,16..23}, lanes>=16 hold K {8..15,24..31}.
__device__ __forceinline__ v16bf load_Bt(const __bf16* base, int stride, int lane) {
  int nn = lane & 15;
  int kb = (lane < 16) ? 0 : 8;
  const __bf16* p = base + nn * stride + kb;
  v8bf lo = *(const v8bf*)(p);
  v8bf hi = *(const v8bf*)(p + 16);
  return __builtin_shufflevector(lo, hi, 0, 1, 2, 3, 4, 5, 6, 7,
                                 8, 9, 10, 11, 12, 13, 14, 15);
}

// ---------------------------------------------------------------------------
// K0: parameter prep: c3w -> bf16; stack 4 branch pointwise convs into one
// 64x64 weight with BN scale folded (Weff bf16, beff f32).
// ---------------------------------------------------------------------------
__global__ void k0_prep(const float* c3w,
                        const float* br_pw, const float* br_pb, const float* br_pg, const float* br_pbeta,
                        const float* mp_pw, const float* mp_pb, const float* mp_g1, const float* mp_b1,
                        const float* p3w, const float* p3b, const float* p3g, const float* p3beta,
                        __bf16* c3wb, __bf16* weff, float* beff) {
  int tid = threadIdx.x;
  const float rs = rsqrtf(1.0f + 1e-5f);
  for (int i = tid; i < S_ * O_ * C_; i += 256) c3wb[i] = f2b(c3w[i]);
  for (int i = tid; i < 64 * 64; i += 256) {
    int oc = i >> 6, c = i & 63;
    float w, g;
    if (oc < 32)      { w = br_pw[oc * 64 + c];        g = br_pg[oc];      }
    else if (oc < 48) { w = mp_pw[(oc - 32) * 64 + c]; g = mp_g1[oc - 32]; }
    else              { w = p3w[(oc - 48) * 64 + c];   g = p3g[oc - 48];   }
    weff[i] = f2b(w * g * rs);
  }
  for (int oc = tid; oc < 64; oc += 256) {
    float b, g, beta;
    if (oc < 32)      { b = br_pb[oc];      g = br_pg[oc];      beta = br_pbeta[oc];   }
    else if (oc < 48) { b = mp_pb[oc - 32]; g = mp_g1[oc - 32]; beta = mp_b1[oc - 32]; }
    else              { b = p3b[oc - 48];   g = p3g[oc - 48];   beta = p3beta[oc - 48];}
    beff[oc] = b * g * rs + beta;
  }
}

// ---------------------------------------------------------------------------
// K1: xm[n,c,v] = mean over T of x[n,c,t,v]
// ---------------------------------------------------------------------------
__global__ void k1_mean(const float* __restrict__ x, float* __restrict__ xm) {
  int n = blockIdx.x;
  for (int idx = threadIdx.x; idx < C_ * V_; idx += 256) {
    int c = idx / V_, v = idx % V_;
    const float* p = x + ((size_t)(n * C_ + c) * T_) * V_ + v;
    float s = 0.f;
#pragma unroll 4
    for (int t = 0; t < T_; ++t) s += p[t * V_];
    xm[n * C_ * V_ + idx] = s * (1.0f / (float)T_);
  }
}

// ---------------------------------------------------------------------------
// K2: channel-wise affinity + softmax.  Output aff[n,s,o][u(32)][v(32)] bf16:
// WMMA-B layout [n=u][k=v] with zero padding (v-pad must be exact zero —
// it is the K-dim zero that makes stage-2's padded K=32 exact).
// ---------------------------------------------------------------------------
__global__ void k2_aff(const float* __restrict__ xm,
                       const float* c1w, const float* c1b,
                       const float* c2w, const float* c2b,
                       const float* c4w, const float* c4b,
                       const float* alpha, const float* PA,
                       __bf16* __restrict__ aff) {
  int n = blockIdx.x / S_;
  int s = blockIdx.x % S_;
  __shared__ float x1l[R_][V_], x2l[R_][V_], dif[R_][V_][V_];
  const float* xmn = xm + n * C_ * V_;

  for (int idx = threadIdx.x; idx < 2 * R_ * V_; idx += 256) {
    int which = idx / (R_ * V_);
    int rv = idx % (R_ * V_);
    int r = rv / V_, v = rv % V_;
    const float* w = (which ? c2w : c1w) + (s * R_ + r) * C_;
    float acc = (which ? c2b : c1b)[s * R_ + r];
#pragma unroll 8
    for (int c = 0; c < C_; ++c) acc += w[c] * xmn[c * V_ + v];
    if (which) x2l[r][v] = acc; else x1l[r][v] = acc;
  }
  __syncthreads();
  for (int idx = threadIdx.x; idx < R_ * V_ * V_; idx += 256) {
    int r = idx / (V_ * V_);
    int uv = idx % (V_ * V_);
    int u = uv / V_, v = uv % V_;
    dif[r][u][v] = tanhf(x1l[r][u] - x2l[r][v]);
  }
  __syncthreads();

  float al = alpha[s];
  __bf16* out = aff + (size_t)(n * S_ + s) * O_ * 1024;
  // softmax over u at fixed (o,v); store row-major [u][v]
  for (int idx = threadIdx.x; idx < O_ * V_; idx += 256) {
    int o = idx / V_, v = idx % V_;
    float a[V_];
    float mx = -1e30f;
#pragma unroll
    for (int u = 0; u < V_; ++u) {
      float acc = c4b[s * O_ + o];
#pragma unroll
      for (int r = 0; r < R_; ++r) acc += c4w[(s * O_ + o) * R_ + r] * dif[r][u][v];
      acc = acc * al + PA[(s * V_ + u) * V_ + v] + PA[(s * V_ + v) * V_ + u];
      a[u] = acc;
      mx = fmaxf(mx, acc);
    }
    float sum = 0.f;
#pragma unroll
    for (int u = 0; u < V_; ++u) { a[u] = expf(a[u] - mx); sum += a[u]; }
    float inv = 1.0f / sum;
#pragma unroll
    for (int u = 0; u < V_; ++u) out[o * 1024 + u * 32 + v] = f2b(a[u] * inv);
#pragma unroll
    for (int u = V_; u < 32; ++u) out[o * 1024 + u * 32 + v] = f2b(0.f);
  }
  // zero K-padding columns v = 25..31 for every u row
  for (int idx = threadIdx.x; idx < O_ * 32 * (32 - V_); idx += 256) {
    int o = idx / (32 * (32 - V_));
    int rest = idx % (32 * (32 - V_));
    int u = rest / (32 - V_);
    int v = V_ + rest % (32 - V_);
    out[o * 1024 + u * 32 + v] = f2b(0.f);
  }
}

// ---------------------------------------------------------------------------
// K3: main GCN.  Per block (n, t-tile of 16):
//  stage1 (WMMA): X3[o, t*32+v] = c3w[s] (64x64) @ Xtile[c, t*32+v]  (+bias)
//  stage2 (WMMA): acc2[k] = X3_o[t,v] @ aff_o[u,v]^T + acc2[k]
//     (16 register-resident C tiles per wave, accumulated over s via the
//      WMMA C operand; X3 v-pad may be garbage since aff v-pad is zero)
//  GCN BatchNorm applied on the register->global store.
// LDS: XtT 64KB + X3 64KB + W 8KB + c3b 768B + bn 512B = 137.5KB dynamic.
// ---------------------------------------------------------------------------
__global__ __launch_bounds__(256, 1)
void k3_gcn(const float* __restrict__ x, const __bf16* __restrict__ c3wb,
            const float* __restrict__ c3b, const __bf16* __restrict__ aff,
            const float* gbn_g, const float* gbn_b, float* __restrict__ yout) {
  int n = blockIdx.x;
  int t0 = blockIdx.y * 16;
  int tid = threadIdx.x, wave = tid >> 5, lane = tid & 31;

  extern __shared__ __align__(16) char smem[];
  __bf16* XtT  = (__bf16*)smem;               // [512][64]  (t*32+v)(c)  B-layout
  __bf16* X3   = (__bf16*)(smem + 65536);     // [64][512]  (o)(t*32+v)  A-layout
  __bf16* Wl   = (__bf16*)(smem + 131072);    // [64][64]
  float*  c3bl = (float*)(smem + 139264);     // [192]
  float*  gsl  = (float*)(smem + 140032);     // [64]
  float*  gbl  = (float*)(smem + 140288);     // [64]

  for (int i = tid; i < 64 * 512; i += 256) {
    int c = i >> 9, col = i & 511, t = col >> 5, v = col & 31;
    float val = (v < V_) ? x[((size_t)(n * C_ + c) * T_ + t0 + t) * V_ + v] : 0.f;
    XtT[col * 64 + c] = f2b(val);
  }
  for (int i = tid; i < S_ * O_; i += 256) c3bl[i] = c3b[i];
  if (tid < O_) {
    gsl[tid] = gbn_g[tid] * rsqrtf(1.0f + 1e-5f);
    gbl[tid] = gbn_b[tid];
  }

  v8f zero8 = {0.f, 0.f, 0.f, 0.f, 0.f, 0.f, 0.f, 0.f};
  v8f acc2[16];
#pragma unroll
  for (int k = 0; k < 16; ++k) acc2[k] = zero8;

  int mb = (lane < 16) ? 0 : 8;

  for (int s = 0; s < S_; ++s) {
    __syncthreads();
    for (int i = tid; i < 64 * 64; i += 256) Wl[i] = c3wb[s * 4096 + i];
    // prefetch this (n,s)'s attention block (128KB) into cache for stage 2
    const __bf16* affn = aff + (size_t)(n * S_ + s) * O_ * 1024;
    for (int off = tid * 128; off < O_ * 1024; off += 256 * 128)
      __builtin_prefetch((const void*)(affn + off), 0, 1);
    __syncthreads();

    // ---- stage 1: 4 o-tiles x 32 n-tiles, K = 64 (2 steps) ----
    for (int tile = wave; tile < 128; tile += 8) {
      int ot = tile >> 5, nt = tile & 31;
      v8f acc = zero8;
#pragma unroll
      for (int ks = 0; ks < 2; ++ks) {
        v16bf a = load_A(Wl + ot * 16 * 64 + ks * 32, 64, lane);
        v16bf b = load_Bt(XtT + nt * 16 * 64 + ks * 32, 64, lane);
        acc = __builtin_amdgcn_wmma_f32_16x16x32_bf16(false, a, false, b,
                                                      (short)0, acc, false, false);
      }
      int col = nt * 16 + (lane & 15);
      float bias[8];
#pragma unroll
      for (int i = 0; i < 8; ++i) bias[i] = c3bl[s * O_ + ot * 16 + mb + i];
      // unconditional, branch-free store (v-pad garbage is harmless: the
      // matching aff K-rows are zero)
#pragma unroll
      for (int i = 0; i < 8; ++i)
        X3[(ot * 16 + mb + i) * 512 + col] = f2b(acc[i] + bias[i]);
    }
    __syncthreads();

    // ---- stage 2: per wave 16 fixed (o, u-tile) tiles, C-accumulated ----
#pragma unroll
    for (int k = 0; k < 16; ++k) {
      int tile = wave + 8 * k;
      int o = tile >> 1, ut = tile & 1;
      v16bf a = load_A(X3 + o * 512, 32, lane);
      v16bf b = load_Bt(affn + o * 1024 + ut * 512, 32, lane);
      acc2[k] = __builtin_amdgcn_wmma_f32_16x16x32_bf16(false, a, false, b,
                                                        (short)0, acc2[k],
                                                        false, false);
    }
  }

  // ---- epilogue: BN + store straight from registers ----
#pragma unroll
  for (int k = 0; k < 16; ++k) {
    int tile = wave + 8 * k;
    int o = tile >> 1, ut = tile & 1;
    int u = ut * 16 + (lane & 15);
    if (u < V_) {
      float gs = gsl[o], gb = gbl[o];
#pragma unroll
      for (int i = 0; i < 8; ++i)
        yout[((size_t)(n * O_ + o) * T_ + t0 + mb + i) * V_ + u] =
            acc2[k][i] * gs + gb;
    }
  }
}

// ---------------------------------------------------------------------------
// K3b: pooled[n,o] = mean over (t,u) of yout  (one block per (n,o))
// ---------------------------------------------------------------------------
__global__ void k3b_pool(const float* __restrict__ y, float* __restrict__ pooled) {
  int no = blockIdx.x;
  __shared__ float red[256];
  const float* p = y + (size_t)no * (T_ * V_);
  float s = 0.f;
  for (int i = threadIdx.x; i < T_ * V_; i += 256) s += p[i];
  red[threadIdx.x] = s;
  __syncthreads();
  for (int off = 128; off > 0; off >>= 1) {
    if (threadIdx.x < off) red[threadIdx.x] += red[threadIdx.x + off];
    __syncthreads();
  }
  if (threadIdx.x == 0) pooled[no] = red[0] / (float)(T_ * V_);
}

// ---------------------------------------------------------------------------
// K4: squeeze-excite: se = sigmoid(relu(pooled@W1^T+b1)@W2^T+b2)
// ---------------------------------------------------------------------------
__global__ void k4_se(const float* __restrict__ pooled,
                      const float* att1_w, const float* att1_b,
                      const float* att2_w, const float* att2_b,
                      float* __restrict__ se) {
  int n = blockIdx.x, tid = threadIdx.x;
  __shared__ float pl[O_], h[AC_];
  if (tid < O_) pl[tid] = pooled[n * O_ + tid];
  __syncthreads();
  if (tid < AC_) {
    float acc = att1_b[tid];
#pragma unroll
    for (int o = 0; o < O_; ++o) acc += att1_w[tid * O_ + o] * pl[o];
    h[tid] = fmaxf(acc, 0.f);
  }
  __syncthreads();
  if (tid < O_) {
    float acc = att2_b[tid];
#pragma unroll
    for (int a = 0; a < AC_; ++a) acc += att2_w[tid * AC_ + a] * h[a];
    se[n * O_ + tid] = 1.0f / (1.0f + expf(-acc));
  }
}

// ---------------------------------------------------------------------------
// K5: z = relu(y*se + x); P = BN(Wall @ z) (+relu for oc<48) via WMMA.
// LDS: zbT 128KB + W 8KB + beff 256B = 136KB dynamic.
// ---------------------------------------------------------------------------
__global__ __launch_bounds__(256, 1)
void k5_branch_pw(const float* __restrict__ y, const float* __restrict__ se,
                  const float* __restrict__ x, const __bf16* __restrict__ weff,
                  const float* __restrict__ beff, __bf16* __restrict__ P) {
  int n = blockIdx.x;
  int t0 = blockIdx.y * 32;
  int tid = threadIdx.x, wave = tid >> 5, lane = tid & 31;
  extern __shared__ __align__(16) char smem[];
  __bf16* zbT   = (__bf16*)smem;             // [1024][64] (t*32+v)(c) B-layout
  __bf16* Wl    = (__bf16*)(smem + 131072);  // [64][64]
  float*  beffl = (float*)(smem + 139264);   // [64]

  for (int i = tid; i < 64 * 64; i += 256) Wl[i] = weff[i];
  if (tid < 64) beffl[tid] = beff[tid];
  for (int i = tid; i < 64 * 1024; i += 256) {
    int c = i >> 10, col = i & 1023, t = col >> 5, v = col & 31;
    float val = 0.f;
    if (v < V_) {
      size_t g = ((size_t)(n * C_ + c) * T_ + t0 + t) * V_ + v;
      val = fmaxf(y[g] * se[n * O_ + c] + x[g], 0.f);
    }
    zbT[col * 64 + c] = f2b(val);
  }
  __syncthreads();

  v8f zero8 = {0.f, 0.f, 0.f, 0.f, 0.f, 0.f, 0.f, 0.f};
  int mb = (lane < 16) ? 0 : 8;
  for (int tile = wave; tile < 256; tile += 8) {
    int ot = tile >> 6, nt = tile & 63;
    v8f acc = zero8;
#pragma unroll
    for (int ks = 0; ks < 2; ++ks) {
      v16bf a = load_A(Wl + ot * 16 * 64 + ks * 32, 64, lane);
      v16bf b = load_Bt(zbT + nt * 16 * 64 + ks * 32, 64, lane);
      acc = __builtin_amdgcn_wmma_f32_16x16x32_bf16(false, a, false, b,
                                                    (short)0, acc, false, false);
    }
    int col = nt * 16 + (lane & 15);
    int t = col >> 5, v = col & 31;
    float bias[8];
#pragma unroll
    for (int i = 0; i < 8; ++i) bias[i] = beffl[ot * 16 + mb + i];
    if (v < V_) {
#pragma unroll
      for (int i = 0; i < 8; ++i) {
        int oc = ot * 16 + mb + i;
        float val = acc[i] + bias[i];
        if (oc < 48) val = fmaxf(val, 0.f);
        P[((size_t)(n * 64 + oc) * T_ + t0 + t) * V_ + v] = f2b(val);
      }
    }
  }
}

// ---------------------------------------------------------------------------
// K6: temporal convs (d=1,2), maxpool3, p3 pass-through; outer residual+relu.
// ---------------------------------------------------------------------------
__global__ void k6_final(const __bf16* __restrict__ P, const float* __restrict__ x,
                         const float* br_tw, const float* br_tb,
                         const float* br_tg, const float* br_tbeta,
                         const float* mp_g2, const float* mp_b2,
                         float* __restrict__ out) {
  int nt = blockIdx.x;
  int n = nt / T_, t = nt % T_;
  const float rs = rsqrtf(1.0f + 1e-5f);
  for (int idx = threadIdx.x; idx < 64 * V_; idx += 256) {
    int c = idx / V_, v = idx % V_;
    float val;
    if (c < 32) {                       // temporal conv branches
      int br = c >> 4, oc = c & 15, d = br + 1;
      float acc = br_tb[br * 16 + oc];
#pragma unroll
      for (int k = 0; k < K_; ++k) {
        int tt2 = t + (k - 2) * d;
        if (tt2 >= 0 && tt2 < T_) {
#pragma unroll
          for (int ic = 0; ic < 16; ++ic) {
            acc += br_tw[((br * 16 + oc) * 16 + ic) * K_ + k] *
                   b2f(P[((size_t)(n * 64 + br * 16 + ic) * T_ + tt2) * V_ + v]);
          }
        }
      }
      val = acc * (br_tg[br * 16 + oc] * rs) + br_tbeta[br * 16 + oc];
    } else if (c < 48) {                // maxpool branch
      float m = -1e30f;
#pragma unroll
      for (int dt = -1; dt <= 1; ++dt) {
        int tt2 = t + dt;
        if (tt2 >= 0 && tt2 < T_)
          m = fmaxf(m, b2f(P[((size_t)(n * 64 + c) * T_ + tt2) * V_ + v]));
      }
      val = m * (mp_g2[c - 32] * rs) + mp_b2[c - 32];
    } else {                            // p3 branch (no relu before concat)
      val = b2f(P[((size_t)(n * 64 + c) * T_ + t) * V_ + v]);
    }
    size_t g = ((size_t)(n * 64 + c) * T_ + t) * V_ + v;
    out[g] = fmaxf(val + x[g], 0.f);
  }
}

// ---------------------------------------------------------------------------
// Workspace layout (bytes, all 256B aligned; total ~199 MB)
// ---------------------------------------------------------------------------
static const size_t OFF_XM   = 0;                          //  819,200 f32
static const size_t OFF_AFF  = 819200;                     // 50,331,648 bf16
static const size_t OFF_Y    = 51150848;                   //104,857,600 f32
static const size_t OFF_POOL = 156008448;                  //     32,768 f32
static const size_t OFF_SE   = 156041216;                  //     32,768 f32
static const size_t OFF_P    = 156073984;                  // 52,428,800 bf16
static const size_t OFF_C3WB = 208502784;                  //     24,576 bf16
static const size_t OFF_WEFF = 208527360;                  //      8,192 bf16
static const size_t OFF_BEFF = 208535552;                  //        256 f32

extern "C" void kernel_launch(void* const* d_in, const int* in_sizes, int n_in,
                              void* d_out, int out_size, void* d_ws, size_t ws_size,
                              hipStream_t stream) {
  const float* x      = (const float*)d_in[0];
  const float* PA     = (const float*)d_in[1];
  const float* alpha  = (const float*)d_in[2];
  const float* c1w    = (const float*)d_in[3];
  const float* c1b    = (const float*)d_in[4];
  const float* c2w    = (const float*)d_in[5];
  const float* c2b    = (const float*)d_in[6];
  const float* c3w    = (const float*)d_in[7];
  const float* c3b    = (const float*)d_in[8];
  const float* c4w    = (const float*)d_in[9];
  const float* c4b    = (const float*)d_in[10];
  const float* gbn_g  = (const float*)d_in[11];
  const float* gbn_b  = (const float*)d_in[12];
  const float* att1_w = (const float*)d_in[13];
  const float* att1_b = (const float*)d_in[14];
  const float* att2_w = (const float*)d_in[15];
  const float* att2_b = (const float*)d_in[16];
  const float* br_pw  = (const float*)d_in[17];
  const float* br_pb  = (const float*)d_in[18];
  const float* br_pg  = (const float*)d_in[19];
  const float* br_pbt = (const float*)d_in[20];
  const float* br_tw  = (const float*)d_in[21];
  const float* br_tb  = (const float*)d_in[22];
  const float* br_tg  = (const float*)d_in[23];
  const float* br_tbt = (const float*)d_in[24];
  const float* mp_pw  = (const float*)d_in[25];
  const float* mp_pb  = (const float*)d_in[26];
  const float* mp_g1  = (const float*)d_in[27];
  const float* mp_b1  = (const float*)d_in[28];
  const float* mp_g2  = (const float*)d_in[29];
  const float* mp_b2  = (const float*)d_in[30];
  const float* p3w    = (const float*)d_in[31];
  const float* p3b    = (const float*)d_in[32];
  const float* p3g    = (const float*)d_in[33];
  const float* p3bt   = (const float*)d_in[34];

  char* ws = (char*)d_ws;
  float*  xm     = (float*)(ws + OFF_XM);
  __bf16* aff    = (__bf16*)(ws + OFF_AFF);
  float*  ybuf   = (float*)(ws + OFF_Y);
  float*  pooled = (float*)(ws + OFF_POOL);
  float*  sev    = (float*)(ws + OFF_SE);
  __bf16* P      = (__bf16*)(ws + OFF_P);
  __bf16* c3wb   = (__bf16*)(ws + OFF_C3WB);
  __bf16* weff   = (__bf16*)(ws + OFF_WEFF);
  float*  beff   = (float*)(ws + OFF_BEFF);

  k0_prep<<<1, 256, 0, stream>>>(c3w, br_pw, br_pb, br_pg, br_pbt,
                                 mp_pw, mp_pb, mp_g1, mp_b1,
                                 p3w, p3b, p3g, p3bt, c3wb, weff, beff);
  k1_mean<<<N_, 256, 0, stream>>>(x, xm);
  k2_aff<<<N_ * S_, 256, 0, stream>>>(xm, c1w, c1b, c2w, c2b, c4w, c4b,
                                      alpha, PA, aff);
  // 137.5KB dynamic LDS per workgroup -> 2 blocks per 320KB WGP
  k3_gcn<<<dim3(N_, T_ / 16), 256, 140544, stream>>>(x, c3wb, c3b, aff,
                                                     gbn_g, gbn_b, ybuf);
  k3b_pool<<<N_ * O_, 256, 0, stream>>>(ybuf, pooled);
  k4_se<<<N_, 64, 0, stream>>>(pooled, att1_w, att1_b, att2_w, att2_b, sev);
  // 136KB dynamic LDS per workgroup
  k5_branch_pw<<<dim3(N_, T_ / 32), 256, 139520, stream>>>(ybuf, sev, x,
                                                           weff, beff, P);
  k6_final<<<N_ * T_, 256, 0, stream>>>(P, x, br_tw, br_tb, br_tg, br_tbt,
                                        mp_g2, mp_b2, (float*)d_out);
}
